// MHSAttention_17051020165521
// MI455X (gfx1250) — matrix operand order repaired and checked by reading, not compiled, not run on hardware
//
#include <hip/hip_runtime.h>
#include <hip/hip_bf16.h>

// ---------------------------------------------------------------------------
// MHSA forward for MI455X (gfx1250): bf16 WMMA, fp32 accumulate, TDM staging.
// B=4, N=2048, D=1024, H=16, Dh=64.  QKV stride = 3*H*Dh = 3072.
// ---------------------------------------------------------------------------

#define B_SZ    4
#define N_SEQ   2048
#define D_MODEL 1024
#define H_NUM   16
#define DH      64
#define QKV_ST  3072           // 3*H*Dh
#define SCALE   0.125f         // 64^-0.5

typedef unsigned short u16;
typedef unsigned int   u32;
typedef __attribute__((ext_vector_type(16))) __bf16 bf16x16;
typedef __attribute__((ext_vector_type(8)))  float  f32x8;
typedef __attribute__((ext_vector_type(4)))  u32    u32x4;
typedef __attribute__((ext_vector_type(8)))  u32    u32x8;

union ABFrag {                 // 32 bytes: 16 bf16 elements (8 VGPRs)
    uint4   u4[2];
    bf16x16 f;
};

__device__ __forceinline__ u16 f32_to_bf16(float x) {
    u32 u = __float_as_uint(x);
    u += 0x7FFFu + ((u >> 16) & 1u);   // round-to-nearest-even
    return (u16)(u >> 16);
}

__device__ __forceinline__ f32x8 zero8() {
    f32x8 v;
#pragma unroll
    for (int i = 0; i < 8; ++i) v[i] = 0.0f;
    return v;
}

#define WMMA_BF16(A, Bm, C) \
    __builtin_amdgcn_wmma_f32_16x16x32_bf16(false, (A), false, (Bm), (short)0, (C), false, false)

// 16-lane max reduction via DPP (no LDS, no waits):
// quad_perm xor1, quad_perm xor2, row_half_mirror, row_mirror.
__device__ __forceinline__ float max16_dpp(float x) {
    int v;
    v = __float_as_int(x);
    x = fmaxf(x, __int_as_float(__builtin_amdgcn_update_dpp(v, v, 0xB1, 0xF, 0xF, true)));
    v = __float_as_int(x);
    x = fmaxf(x, __int_as_float(__builtin_amdgcn_update_dpp(v, v, 0x4E, 0xF, 0xF, true)));
    v = __float_as_int(x);
    x = fmaxf(x, __int_as_float(__builtin_amdgcn_update_dpp(v, v, 0x141, 0xF, 0xF, true)));
    v = __float_as_int(x);
    x = fmaxf(x, __int_as_float(__builtin_amdgcn_update_dpp(v, v, 0x140, 0xF, 0xF, true)));
    return x;
}

// ---------------------------------------------------------------------------
// Tensor Data Mover: 2D tile (tileW x tileH elements of 2 bytes) from global
// row-major (rowStride elements) into contiguous LDS at byte offset ldsOff.
// Descriptor bit layout per CDNA5 ISA ch.8 (group0 128b, group1 256b).
// ---------------------------------------------------------------------------
__device__ __forceinline__ void tdm_load_2d_bf16(u32 ldsOff, const void* gaddr,
                                                 u32 tileW, u32 tileH, u32 rowStride) {
    unsigned long long ga = (unsigned long long)gaddr;
    u32x4 g0;
    g0[0] = 1u;                                        // count=1, user descriptor
    g0[1] = ldsOff;                                    // lds_addr (bytes)
    g0[2] = (u32)ga;                                   // global_addr[31:0]
    g0[3] = (u32)(ga >> 32) | (2u << 30);              // global_addr[56:32] | type=2
    u32x8 g1;
    g1[0] = 1u << 16;                                  // data_size=1 (2 bytes)
    g1[1] = (tileW & 0xFFFFu) << 16;                   // tensor_dim0[15:0]
    g1[2] = ((tileW >> 16) & 0xFFFFu) | ((tileH & 0xFFFFu) << 16); // dim0 hi | tensor_dim1 lo
    g1[3] = ((tileH >> 16) & 0xFFFFu) | ((tileW & 0xFFFFu) << 16); // dim1 hi | tile_dim0
    g1[4] = tileH & 0xFFFFu;                           // tile_dim1 (tile_dim2=0)
    g1[5] = rowStride;                                 // tensor_dim0_stride[31:0]
    g1[6] = 0u;                                        // stride hi | tensor_dim1_stride lo
    g1[7] = 0u;
    asm volatile("tensor_load_to_lds %0, %1" :: "s"(g0), "s"(g1) : "memory");
}

// ---------------------------------------------------------------------------
// Pack kernels
// ---------------------------------------------------------------------------
__global__ void pack_bf16(const float* __restrict__ in, u16* __restrict__ out, int n) {
    int i = blockIdx.x * blockDim.x + threadIdx.x;
    if (i < n) out[i] = f32_to_bf16(in[i]);
}

// in [R,C] fp32 -> out [C,R] bf16
__global__ void transpose_to_bf16(const float* __restrict__ in, u16* __restrict__ out,
                                  int R, int C) {
    int c = blockIdx.x * 16 + threadIdx.x;
    int r = blockIdx.y * 16 + threadIdx.y;
    out[(size_t)c * R + r] = f32_to_bf16(in[(size_t)r * C + c]);
}

// vT[((b*H+h)*Dh+d)*N + n] = qkv[(b*N+n)*3072 + 2048 + h*64 + d]
__global__ void pack_vT(const u16* __restrict__ qkv, u16* __restrict__ vT) {
    int i = blockIdx.x * blockDim.x + threadIdx.x;          // < B*H*Dh*N
    int n = i & (N_SEQ - 1);
    int d = (i >> 11) & (DH - 1);
    int h = (i >> 17) & (H_NUM - 1);
    int b = i >> 21;
    vT[i] = qkv[((size_t)(b * N_SEQ + n)) * QKV_ST + 2 * H_NUM * DH + h * DH + d];
}

// ---------------------------------------------------------------------------
// GEMM: C[M,Nc] = A[M,K] * Bt[Nc,K]^T.  One wave: 16x128 strip (8 C tiles,
// A-fragment reused 8x).  Block = 8 waves -> 16 rows x 1024 cols.
// Requires M%16==0, Nc%1024==0, K%32==0.
// ---------------------------------------------------------------------------
template <bool OUT_BF16>
__global__ __launch_bounds__(256, 1)
void gemm_bf16_wmma(const u16* __restrict__ A, const u16* __restrict__ Bt,
                    void* __restrict__ C, int M, int Nc, int K) {
    const int lane = threadIdx.x & 31;
    const int wid  = threadIdx.x >> 5;
    const int half = lane >> 4;
    const int l16  = lane & 15;

    const int rowA = blockIdx.y * 16 + l16;
    const int col0 = blockIdx.x * 1024 + wid * 128;

    f32x8 acc[8];
#pragma unroll
    for (int t = 0; t < 8; ++t) acc[t] = zero8();

    const u16* aRow = A + (size_t)rowA * K;
    for (int k0 = 0; k0 < K; k0 += 32) {
        ABFrag a;
        const u16* ap = aRow + k0 + half * 8;               // e0..7 -> K=k0+half*8+e
        a.u4[0] = *(const uint4*)(ap);
        a.u4[1] = *(const uint4*)(ap + 16);                 // e8..15 -> K=k0+16+half*8+e
        __builtin_prefetch(ap + 128, 0, 1);
#pragma unroll
        for (int t = 0; t < 8; ++t) {
            const int col = col0 + t * 16 + l16;
            ABFrag bm;
            const u16* bp = Bt + (size_t)col * K + k0 + half * 8;  // K = half*8 + e, e8..15 -> 16+half*8+e
            bm.u4[0] = *(const uint4*)(bp);
            bm.u4[1] = *(const uint4*)(bp + 16);
            acc[t] = WMMA_BF16(a.f, bm.f, acc[t]);
        }
    }

#pragma unroll
    for (int t = 0; t < 8; ++t) {
#pragma unroll
        for (int r = 0; r < 8; ++r) {
            const int ro = blockIdx.y * 16 + r + half * 8;  // C layout: M = r + 8*half
            const int co = col0 + t * 16 + l16;
            if (OUT_BF16)
                ((u16*)C)[(size_t)ro * Nc + co] = f32_to_bf16(acc[t][r]);
            else
                ((float*)C)[(size_t)ro * Nc + co] = acc[t][r];
        }
    }
}

// ---------------------------------------------------------------------------
// Flash attention, one (b,h) per block, 8 waves x 16 query rows.
// TDM double-buffers the 32-key K/V tiles into LDS (shared by all 8 waves);
// per step: 4 WMMA for S, DPP row-max, 5 WMMA for O += P*V and l += P*ones.
// Dynamic LDS layout (bytes):
//   [0,8192)      K tiles  [2][32 keys][64 dh] bf16
//   [8192,16384)  V tiles  [2][64 dh][32 keys] bf16
//   [16384,24576) P tiles  [8 waves][16][32] bf16
// ---------------------------------------------------------------------------
__global__ __launch_bounds__(256, 1)
void flash_attn(const u16* __restrict__ qkv, const u16* __restrict__ vT,
                u16* __restrict__ attn_out) {
    extern __shared__ char smem[];
    const int lane = threadIdx.x & 31;
    const int wid  = threadIdx.x >> 5;
    const int half = lane >> 4;
    const int l16  = lane & 15;

    const int b = blockIdx.z, h = blockIdx.y;
    const int qbase = blockIdx.x * 128 + wid * 16;

    u16* myP = (u16*)(smem + 16384 + wid * 1024);

    // Q fragments (16x32 each), dh slabs [0,32) / [32,64)
    ABFrag qa[2];
    {
        const u16* qp = qkv + ((size_t)(b * N_SEQ + qbase + l16)) * QKV_ST + h * DH;
#pragma unroll
        for (int s = 0; s < 2; ++s) {
            const u16* p = qp + s * 32 + half * 8;
            qa[s].u4[0] = *(const uint4*)(p);
            qa[s].u4[1] = *(const uint4*)(p + 16);
        }
    }

    // ones B-fragment (32x16 of bf16 1.0) for row-sum WMMA
    ABFrag ones;
#pragma unroll
    for (int i = 0; i < 2; ++i) {
        ones.u4[i].x = 0x3F803F80u; ones.u4[i].y = 0x3F803F80u;
        ones.u4[i].z = 0x3F803F80u; ones.u4[i].w = 0x3F803F80u;
    }

    f32x8 o[4];
#pragma unroll
    for (int t = 0; t < 4; ++t) o[t] = zero8();
    f32x8 ls = zero8();                    // running row sums (C layout, replicated)
    float mrow[8];
#pragma unroll
    for (int r = 0; r < 8; ++r) mrow[r] = -3.0e38f;

    const u16* kbase = qkv + (size_t)b * N_SEQ * QKV_ST + H_NUM * DH + h * DH;
    const u16* vTb   = vT + ((size_t)((b * H_NUM + h) * DH)) * N_SEQ;

    // Prime buffer 0 via TDM (wave 0 drives the DMA for the whole block).
    if (wid == 0) {
        tdm_load_2d_bf16(0u,    kbase, DH, 32, QKV_ST);           // K rows j=0..31
        tdm_load_2d_bf16(8192u, vTb,   32, DH, N_SEQ);            // V^T cols j=0..31
        asm volatile("s_wait_tensorcnt 0x0" ::: "memory");
    }
    __syncthreads();

    for (int j = 0; j < N_SEQ; j += 32) {
        const int cur = (j >> 5) & 1;
        const u16* Kc = (const u16*)(smem + cur * 4096);
        const u16* Vc = (const u16*)(smem + 8192 + cur * 4096);

        // Kick off DMA for the next 32-key tile into the other buffer.
        if (wid == 0 && j + 32 < N_SEQ) {
            const int nxt = cur ^ 1;
            tdm_load_2d_bf16((u32)(nxt * 4096),
                             kbase + (size_t)(j + 32) * QKV_ST, DH, 32, QKV_ST);
            tdm_load_2d_bf16((u32)(8192 + nxt * 4096),
                             vTb + (j + 32), 32, DH, N_SEQ);
        }

        // ---- S = Q @ K^T : two 16x16 tiles from LDS K ----
        f32x8 stile[2];
#pragma unroll
        for (int t = 0; t < 2; ++t) {
            f32x8 sacc = zero8();
#pragma unroll
            for (int s = 0; s < 2; ++s) {
                ABFrag kb;
                const u16* p = Kc + (t * 16 + l16) * DH + s * 32 + half * 8;
                kb.u4[0] = *(const uint4*)(p);
                kb.u4[1] = *(const uint4*)(p + 16);
                sacc = WMMA_BF16(qa[s].f, kb.f, sacc);
            }
            stile[t] = sacc;
        }

        // ---- online softmax stats (row = r + 8*half), DPP max reduce ----
#pragma unroll
        for (int r = 0; r < 8; ++r) {
            const float slo = stile[0][r] * SCALE;
            const float shi = stile[1][r] * SCALE;
            const float mx = max16_dpp(fmaxf(slo, shi));
            const float mnew = fmaxf(mrow[r], mx);
            const float corr = __expf(mrow[r] - mnew);
            mrow[r] = mnew;
            ls[r] *= corr;
#pragma unroll
            for (int t = 0; t < 4; ++t) o[t][r] *= corr;
            const int prow = r + half * 8;
            myP[prow * 32 + l16]      = f32_to_bf16(__expf(slo - mnew));
            myP[prow * 32 + 16 + l16] = f32_to_bf16(__expf(shi - mnew));
        }

        asm volatile("s_wait_dscnt 0x0" ::: "memory");  // P stores visible to wave

        // ---- P A-fragment (16x32) from LDS ----
        ABFrag pa;
        {
            const u16* pr = myP + l16 * 32 + half * 8;
            pa.u4[0] = *(const uint4*)(pr);
            pa.u4[1] = *(const uint4*)(pr + 16);
        }

        // ---- O += P @ V (four dh tiles) ; l += P @ ones ----
#pragma unroll
        for (int t = 0; t < 4; ++t) {
            ABFrag vb;
            const u16* vp = Vc + (t * 16 + l16) * 32 + half * 8;
            vb.u4[0] = *(const uint4*)(vp);
            vb.u4[1] = *(const uint4*)(vp + 16);
            o[t] = WMMA_BF16(pa.f, vb.f, o[t]);
        }
        ls = WMMA_BF16(pa.f, ones.f, ls);

        // Next tile must be resident and everyone done with 'cur' before reuse.
        if (wid == 0 && j + 32 < N_SEQ)
            asm volatile("s_wait_tensorcnt 0x0" ::: "memory");
        __syncthreads();
    }

    // ---- normalize and store bf16 [B,N,H*Dh] ----
#pragma unroll
    for (int r = 0; r < 8; ++r) {
        const float inv = 1.0f / ls[r];
        const int qrow = qbase + r + half * 8;
        u16* op = attn_out + ((size_t)(b * N_SEQ + qrow)) * (H_NUM * DH) + h * DH;
#pragma unroll
        for (int t = 0; t < 4; ++t)
            op[t * 16 + l16] = f32_to_bf16(o[t][r] * inv);
    }
}

// ---------------------------------------------------------------------------
// Launcher
// ---------------------------------------------------------------------------
extern "C" void kernel_launch(void* const* d_in, const int* in_sizes, int n_in,
                              void* d_out, int out_size, void* d_ws, size_t ws_size,
                              hipStream_t stream) {
    const float* x     = (const float*)d_in[0];   // [B,N,D]
    const float* w_qkv = (const float*)d_in[1];   // [D, 3072]
    const float* w_out = (const float*)d_in[2];   // [1024, 1024]
    float* out = (float*)d_out;                   // [B,N,D]

    const size_t MN = (size_t)B_SZ * N_SEQ;       // 8192
    u16* ws = (u16*)d_ws;
    size_t off = 0;
    u16* xb     = ws + off; off += MN * D_MODEL;
    u16* wqkvT  = ws + off; off += (size_t)QKV_ST * D_MODEL;
    u16* woutT  = ws + off; off += (size_t)D_MODEL * D_MODEL;
    u16* qkvb   = ws + off; off += MN * QKV_ST;
    u16* vTb    = ws + off; off += MN * D_MODEL;
    u16* attnb  = ws + off; off += MN * D_MODEL;
    (void)off; (void)ws_size; (void)in_sizes; (void)n_in; (void)out_size;

    {   // 1) pack x -> bf16
        int n = (int)(MN * D_MODEL);
        pack_bf16<<<n / 256, 256, 0, stream>>>(x, xb, n);
    }
    // 2) transpose+pack weights
    transpose_to_bf16<<<dim3(QKV_ST / 16, D_MODEL / 16), dim3(16, 16), 0, stream>>>(
        w_qkv, wqkvT, D_MODEL, QKV_ST);
    transpose_to_bf16<<<dim3(D_MODEL / 16, D_MODEL / 16), dim3(16, 16), 0, stream>>>(
        w_out, woutT, D_MODEL, D_MODEL);
    // 3) QKV projection -> bf16 [8192,3072]
    gemm_bf16_wmma<true><<<dim3(QKV_ST / 1024, (int)(MN / 16)), 256, 0, stream>>>(
        xb, wqkvT, qkvb, (int)MN, QKV_ST, D_MODEL);
    {   // 4) pack V^T per head
        int n = (int)(MN * D_MODEL);
        pack_vT<<<n / 256, 256, 0, stream>>>(qkvb, vTb);
    }
    // 5) flash attention (TDM-staged K/V) -> attnb bf16
    flash_attn<<<dim3(N_SEQ / 128, H_NUM, B_SZ), 256, 24576, stream>>>(qkvb, vTb, attnb);
    // 6) output projection -> fp32 d_out
    gemm_bf16_wmma<false><<<dim3(D_MODEL / 1024, (int)(MN / 16)), 256, 0, stream>>>(
        attnb, woutT, out, (int)MN, D_MODEL, D_MODEL);
}
